// VectorNet_20933670601302
// MI455X (gfx1250) — matrix operand order repaired and checked by reading, not compiled, not run on hardware
//
#include <hip/hip_runtime.h>
#include <hip/hip_bf16.h>

// ---------------- problem constants (match reference) ----------------
#define B_SZ   32
#define NPOLY  256
#define VPP    64
#define LCH    32
#define VROWS  (1 + NPOLY * VPP)   // 16385
#define PLEN   256                 // L * 2^3
#define MDIM   128                 // compact P = [m, m], m has 128 channels
#define LN_EPS 1e-5f

typedef __attribute__((ext_vector_type(16))) _Float16 v16h;
typedef __attribute__((ext_vector_type(8)))  _Float16 v8h;
typedef __attribute__((ext_vector_type(8)))  float    v8f;

// ds_swizzle_b32 group-of-32 XOR exchange: new_lane = lane ^ MASK.
// imm encoding: offset[15]=0, and_mask=0x1f, or_mask=0, xor_mask=MASK.
template<int MASK>
__device__ __forceinline__ float swz_xor(float x) {
    return __int_as_float(
        __builtin_amdgcn_ds_swizzle(__float_as_int(x), (MASK << 10) | 0x1f));
}

// =====================================================================
// One encoder layer, fully specialized at compile time on C (= c_in = c_out
// of the matmul). 4 waves; wave w owns rows [16w,16w+16).
//   h = relu(LN(x @ W + b) * g + beta);  agg = colmax(h)
//   if !FINAL: x <- [h | agg broadcast]  (width 2C)
//   if  FINAL: aggrow holds m = colmax(h)
// =====================================================================
template<int C, bool FINAL>
__device__ __forceinline__ void vn_layer(
    const int tid, const int w, const int mrow, const int g16,
    _Float16 (* __restrict__ xh)[128], _Float16* __restrict__ whT,
    float (* __restrict__ aggbuf)[128], float* __restrict__ aggrow,
    const float* __restrict__ W,     const float* __restrict__ bias,
    const float* __restrict__ gamma, const float* __restrict__ beta)
{
    constexpr int NT = C / 16;                    // N tiles
    constexpr int KS = C / 32;                    // K steps of 32
    constexpr int LC = (C == 32) ? 5 : (C == 64) ? 6 : 7;

    // ---- stage weights transposed into LDS as f16: whT[n*128 + k] ----
    for (int idx = tid; idx < C * C; idx += 128) {
        int k = idx >> LC, n = idx & (C - 1);
        whT[n * 128 + k] = (_Float16)W[idx];
    }
    __syncthreads();

    // ---- A fragments: two contiguous 16B runs per K-step ----
    // 16-bit A layout: elem (2q+h): q<4 -> K=k0+2q+h+8g ; q>=4 -> K=k0+16+2(q-4)+h+8g
    v16h afrag[KS];
    {
        const _Float16* X = &xh[16 * w + mrow][0];
#pragma unroll
        for (int ks = 0; ks < KS; ++ks) {
            const int k0 = ks * 32 + 8 * g16;
            v8h lo = *(const v8h*)(X + k0);        // elems 0..7
            v8h hi = *(const v8h*)(X + k0 + 16);   // elems 8..15
            afrag[ks] = __builtin_shufflevector(lo, hi,
                0,1,2,3,4,5,6,7,8,9,10,11,12,13,14,15);
        }
    }

    // ---- WMMA matmul: acc[nt] = X(16 x C) @ W(C x 16) ----
    v8f acc[NT];
#pragma unroll
    for (int nt = 0; nt < NT; ++nt) {
        v8f cacc = {};
        // B layout: elem (2q+h): q<4 -> K = ks*32 + 2q + h + 8g ; q>=4 -> K = ks*32 + 16 + 2(q-4) + h + 8g, N = nt*16 + mrow
        const _Float16* Wp = &whT[(nt * 16 + mrow) * 128 + 8 * g16];
#pragma unroll
        for (int ks = 0; ks < KS; ++ks) {
            v8h blo = *(const v8h*)(Wp + ks * 32);
            v8h bhi = *(const v8h*)(Wp + ks * 32 + 16);
            v16h bfrag = __builtin_shufflevector(blo, bhi,
                0,1,2,3,4,5,6,7,8,9,10,11,12,13,14,15);
            cacc = __builtin_amdgcn_wmma_f32_16x16x32_f16(
                false, afrag[ks], false, bfrag, (short)0, cacc, false, false);
        }
        acc[nt] = cacc;
    }

    // ---- bias (pre-LN) ----
#pragma unroll
    for (int nt = 0; nt < NT; ++nt) {
        float bv = bias[nt * 16 + mrow];
#pragma unroll
        for (int i = 0; i < 8; ++i) acc[nt][i] += bv;
    }

    // ---- LayerNorm stats fully in registers ----
    // D layout: comp i of lane-half g holds row M = i + 8g; the row's values
    // per N-tile live across the 16 lanes of the half -> ds_swizzle butterfly.
    float mu[8], inv[8];
#pragma unroll
    for (int i = 0; i < 8; ++i) {
        float s = 0.f, s2 = 0.f;
#pragma unroll
        for (int nt = 0; nt < NT; ++nt) {
            float x = acc[nt][i];
            s += x; s2 += x * x;
        }
        s += swz_xor<1>(s);  s += swz_xor<2>(s);
        s += swz_xor<4>(s);  s += swz_xor<8>(s);
        s2 += swz_xor<1>(s2); s2 += swz_xor<2>(s2);
        s2 += swz_xor<4>(s2); s2 += swz_xor<8>(s2);
        float mean = s * (1.0f / C);
        float var  = s2 * (1.0f / C) - mean * mean;
        mu[i]  = mean;
        inv[i] = rsqrtf(var + LN_EPS);
    }

    // ---- gamma/beta, relu, write h + per-wave column max ----
#pragma unroll
    for (int nt = 0; nt < NT; ++nt) {
        const int n = nt * 16 + mrow;
        float gm = gamma[n], bt = beta[n];
        float cm = 0.0f;                           // relu => colmax >= 0
        float hv[8];
#pragma unroll
        for (int i = 0; i < 8; ++i) {
            float x = (acc[nt][i] - mu[i]) * inv[i] * gm + bt;
            x = fmaxf(x, 0.0f);
            hv[i] = x;
            cm = fmaxf(cm, x);
        }
        cm = fmaxf(cm, swz_xor<16>(cm));           // merge lane halves
        if (!FINAL) {
#pragma unroll
            for (int i = 0; i < 8; ++i)
                xh[16 * w + i + 8 * g16][n] = (_Float16)hv[i];
        }
        if (g16 == 0) aggbuf[w][n] = cm;
    }
    __syncthreads();

    // ---- combine agg across the 4 waves ----
    for (int idx = tid; idx < C; idx += 128) {
        aggrow[idx] = fmaxf(fmaxf(aggbuf[0][idx], aggbuf[1][idx]),
                            fmaxf(aggbuf[2][idx], aggbuf[3][idx]));
    }
    __syncthreads();

    if (!FINAL) {
        // x_next = [h | broadcast(agg)] : replicate agg into cols [C, 2C)
        for (int idx = tid; idx < VPP * C; idx += 128) {
            int r = idx >> LC, n = idx & (C - 1);
            xh[r][C + n] = (_Float16)aggrow[n];
        }
        __syncthreads();
    }
}

// =====================================================================
// Kernel 1: fused 3-layer polyline encoder.
// grid = B*NPOLY blocks, 128 threads (4 waves).
// Produces m[b][p][0:128] = colmax of layer-2 h  (P = concat(m, m)).
// =====================================================================
__global__ __launch_bounds__(128) void vn_encoder(
    const float* __restrict__ data,
    const float* __restrict__ W0, const float* __restrict__ b0,
    const float* __restrict__ g0, const float* __restrict__ e0,
    const float* __restrict__ W1, const float* __restrict__ b1,
    const float* __restrict__ g1, const float* __restrict__ e1,
    const float* __restrict__ W2, const float* __restrict__ b2,
    const float* __restrict__ g2, const float* __restrict__ e2,
    float* __restrict__ mOut)
{
    __shared__ _Float16 xh[VPP][128];        // activations, f16   (16 KB)
    __shared__ _Float16 whT[128 * 128];      // weights transposed (32 KB)
    __shared__ float    aggbuf[4][128];      // per-wave column max
    __shared__ float    aggrow[128];         // combined column max

    const int tid  = threadIdx.x;
    const int lane = tid & 31;
    const int w    = tid >> 5;               // wave id == M-tile
    const int blk  = blockIdx.x;
    const int b    = blk >> 8;               // / NPOLY
    const int p    = blk & (NPOLY - 1);

    const int mrow = lane & 15;
    const int g16  = lane >> 4;

    // ---- load x0: rows [b, 1 + p*64 .. +64), last channel zeroed ----
    const float* src = data + ((size_t)b * VROWS + 1 + (size_t)p * VPP) * LCH;
    for (int idx = tid; idx < VPP * LCH; idx += 128) {
        int r = idx >> 5, ch = idx & 31;
        float v = (ch == LCH - 1) ? 0.0f : src[r * LCH + ch];
        xh[r][ch] = (_Float16)v;
    }
    __syncthreads();

    vn_layer< 32, false>(tid, w, mrow, g16, xh, whT, aggbuf, aggrow, W0, b0, g0, e0);
    vn_layer< 64, false>(tid, w, mrow, g16, xh, whT, aggbuf, aggrow, W1, b1, g1, e1);
    vn_layer<128, true >(tid, w, mrow, g16, xh, whT, aggbuf, aggrow, W2, b2, g2, e2);

    float* dst = mOut + (size_t)(b * NPOLY + p) * MDIM;
    for (int idx = tid; idx < MDIM; idx += 128) dst[idx] = aggrow[idx];
}

// =====================================================================
// Kernel 2: attention, reduced to matvecs.
//   q = [m_a,m_a] @ Wq + bq
//   scores_p = m_p . ueff + q.bk,  ueff_j = (Wk q)_j + (Wk q)_{j+128}
//   out = (att . m) @ (Wv[0:128]+Wv[128:256]) + bv
// grid = B blocks, 256 threads.
// =====================================================================
__global__ __launch_bounds__(256) void vn_attn(
    const float* __restrict__ data,
    const float* __restrict__ Wq, const float* __restrict__ bq,
    const float* __restrict__ Wk, const float* __restrict__ bk,
    const float* __restrict__ Wv, const float* __restrict__ bv,
    const float* __restrict__ m,  float* __restrict__ out)
{
    __shared__ float sq[256];
    __shared__ float su[256];
    __shared__ float satt[256];
    __shared__ float sw[128];
    __shared__ float red[256];

    const int b = blockIdx.x, t = threadIdx.x;
    const int aid = (int)data[(size_t)b * VROWS * LCH];   // data[b,0,0]
    const float* mb = m + (size_t)b * NPOLY * MDIM;
    const float* ma = mb + (size_t)aid * MDIM;

    // q[t]
    {
        float acc = bq[t];
        for (int j = 0; j < MDIM; ++j)
            acc += ma[j] * (Wq[j * PLEN + t] + Wq[(j + MDIM) * PLEN + t]);
        sq[t] = acc;
    }
    __syncthreads();

    // u[t] = row t of Wk dot q ; partial of q.bk
    {
        float acc = 0.f;
        const float* row = Wk + (size_t)t * PLEN;
        for (int cI = 0; cI < PLEN; ++cI) acc += row[cI] * sq[cI];
        su[t] = acc;
    }
    red[t] = sq[t] * bk[t];
    __syncthreads();
    for (int s = 128; s > 0; s >>= 1) {
        if (t < s) red[t] += red[t + s];
        __syncthreads();
    }
    const float qbk = red[0];
    if (t < MDIM) sw[t] = su[t] + su[t + MDIM];          // ueff (temp in sw)
    __syncthreads();

    // scores
    {
        const float* mp = mb + (size_t)t * MDIM;
        float acc = qbk;
        for (int j = 0; j < MDIM; ++j) acc += mp[j] * sw[j];
        satt[t] = acc * (1.0f / 16.0f);                  // 1/sqrt(PLEN)
    }
    __syncthreads();

    // softmax over 256
    red[t] = satt[t]; __syncthreads();
    for (int s = 128; s > 0; s >>= 1) {
        if (t < s) red[t] = fmaxf(red[t], red[t + s]);
        __syncthreads();
    }
    const float mx = red[0]; __syncthreads();
    const float e = expf(satt[t] - mx);
    red[t] = e; __syncthreads();
    for (int s = 128; s > 0; s >>= 1) {
        if (t < s) red[t] += red[t + s];
        __syncthreads();
    }
    const float isum = 1.0f / red[0];
    __syncthreads();
    satt[t] = e * isum;
    __syncthreads();

    // w_j = sum_p att_p * m[p][j]
    if (t < MDIM) {
        float acc = 0.f;
        for (int pI = 0; pI < NPOLY; ++pI) acc += satt[pI] * mb[(size_t)pI * MDIM + t];
        sw[t] = acc;
    }
    __syncthreads();

    // out[t]
    {
        float acc = bv[t];
        for (int j = 0; j < MDIM; ++j)
            acc += sw[j] * (Wv[j * PLEN + t] + Wv[(j + MDIM) * PLEN + t]);
        out[(size_t)b * PLEN + t] = acc;
    }
}

// =====================================================================
extern "C" void kernel_launch(void* const* d_in, const int* in_sizes, int n_in,
                              void* d_out, int out_size, void* d_ws, size_t ws_size,
                              hipStream_t stream)
{
    (void)in_sizes; (void)n_in; (void)out_size; (void)ws_size;
    const float* data = (const float*)d_in[0];
    const float* W0 = (const float*)d_in[1];
    const float* b0 = (const float*)d_in[2];
    const float* g0 = (const float*)d_in[3];
    const float* e0 = (const float*)d_in[4];
    const float* W1 = (const float*)d_in[5];
    const float* b1 = (const float*)d_in[6];
    const float* g1 = (const float*)d_in[7];
    const float* e1 = (const float*)d_in[8];
    const float* W2 = (const float*)d_in[9];
    const float* b2 = (const float*)d_in[10];
    const float* g2 = (const float*)d_in[11];
    const float* e2 = (const float*)d_in[12];
    const float* Wq = (const float*)d_in[13];
    const float* bq = (const float*)d_in[14];
    const float* Wk = (const float*)d_in[15];
    const float* bk = (const float*)d_in[16];
    const float* Wv = (const float*)d_in[17];
    const float* bv = (const float*)d_in[18];

    float* mbuf = (float*)d_ws;   // B*NPOLY*128 floats = 4 MB

    vn_encoder<<<B_SZ * NPOLY, 128, 0, stream>>>(
        data, W0, b0, g0, e0, W1, b1, g1, e1, W2, b2, g2, e2, mbuf);

    vn_attn<<<B_SZ, 256, 0, stream>>>(
        data, Wq, bq, Wk, bk, Wv, bv, mbuf, (float*)d_out);
}
